// SelfAttention_87900800680739
// MI455X (gfx1250) — matrix operand order repaired and checked by reading, not compiled, not run on hardware
//
#include <hip/hip_runtime.h>
#include <math.h>

typedef __attribute__((ext_vector_type(16))) __bf16 v16bf;
typedef __attribute__((ext_vector_type(8)))  float  v8f;

union V16 { v16bf v; unsigned short s[16]; uint4 q[2]; };
union V8  { v8f v; float f[8]; };

__device__ __forceinline__ unsigned f2bf1(float x) {
  unsigned u = __builtin_bit_cast(unsigned, x);
  return (u + 0x7fffu + ((u >> 16) & 1u)) >> 16;   // round-to-nearest-even
}
__device__ __forceinline__ unsigned f2bf2(float lo, float hi) {
  return f2bf1(lo) | (f2bf1(hi) << 16);
}
__device__ __forceinline__ unsigned short f2bf(float x) { return (unsigned short)f2bf1(x); }

__device__ __forceinline__ v8f wmma_bf16(v16bf a, v16bf b, v8f c) {
  return __builtin_amdgcn_wmma_f32_16x16x32_bf16(false, a, false, b, (short)0, c, false, false);
}

constexpr int kB = 4, kS = 1024, kD = 4096, kH = 32, kKVH = 8, kHD = 128;
constexpr int kMAXS = 4096, kNREP = kH / kKVH;
constexpr float kLog2_10000 = 13.287712379549449f;

// ---------------------------------------------------------------------------
// Tiled GEMM: C[M,N] = A[M,K] (fp32 row-major) * B[K,N] (fp32 row-major).
// 128x128 block, BK=32, 8 waves, bf16 LDS tiles, per-wave 16(M)x128(N).
// ROPE=true applies rotary embedding to the output in-register (cols = heads
// of 128, pair partner fetched from the adjacent lane with shfl_xor).
// ---------------------------------------------------------------------------
template <bool ROPE>
__global__ __launch_bounds__(256) void gemm_xw_bf16(const float* __restrict__ A,
                                                    const float* __restrict__ B,
                                                    float* __restrict__ C,
                                                    int M, int N, int K,
                                                    const int* __restrict__ sp) {
  __shared__ __align__(16) unsigned short sA[128 * 40];   // [m][k], stride 40
  __shared__ __align__(16) unsigned short sB[128 * 40];   // [n][k] (transposed)
  const int t = threadIdx.x;
  const int wave = t >> 5, lane = t & 31;
  const int hf = lane >> 4, l15 = lane & 15;
  const int m0 = blockIdx.y * 128, n0 = blockIdx.x * 128;
  v8f acc[8] = {};
  const int arow = wave * 16 + l15;
  for (int k0 = 0; k0 < K; k0 += 32) {
    __syncthreads();
#pragma unroll
    for (int i = 0; i < 4; ++i) {
      int qi = t + i * 256;                 // 0..1023
      int row = qi >> 3, kc = (qi & 7) * 4; // A tile: 128 rows x 32 k
      float4 va = *(const float4*)(A + (m0 + row) * K + k0 + kc);
      uint2 u;
      u.x = f2bf2(va.x, va.y);
      u.y = f2bf2(va.z, va.w);
      *(uint2*)(&sA[row * 40 + kc]) = u;
      int kr = qi >> 5, nc = (qi & 31) * 4; // B tile: 32 k x 128 n, transpose
      float4 vb = *(const float4*)(B + (k0 + kr) * N + n0 + nc);
      sB[(nc + 0) * 40 + kr] = f2bf(vb.x);
      sB[(nc + 1) * 40 + kr] = f2bf(vb.y);
      sB[(nc + 2) * 40 + kr] = f2bf(vb.z);
      sB[(nc + 3) * 40 + kr] = f2bf(vb.w);
    }
    __syncthreads();
    // A frag (16x32): lane m=l15, elems 0..7 = k hf*8+0..7, 8..15 = 16+hf*8+..
    V16 af;
    af.q[0] = *(const uint4*)(&sA[arow * 40 + hf * 8]);
    af.q[1] = *(const uint4*)(&sA[arow * 40 + 16 + hf * 8]);
    // Preload all 8 B frags so ds_loads pipeline, then WMMAs go back-to-back.
    V16 bf[8];
#pragma unroll
    for (int j = 0; j < 8; ++j) {
      const int brow = j * 16 + l15;
      bf[j].q[0] = *(const uint4*)(&sB[brow * 40 + hf * 8]);
      bf[j].q[1] = *(const uint4*)(&sB[brow * 40 + 16 + hf * 8]);
    }
#pragma unroll
    for (int j = 0; j < 8; ++j) acc[j] = wmma_bf16(af.v, bf[j].v, acc[j]);
  }
  const int start = ROPE ? sp[0] : 0;
#pragma unroll
  for (int j = 0; j < 8; ++j) {
    V8 cv; cv.v = acc[j];
    const int col = n0 + j * 16 + l15;
    float inv = 0.0f;
    if (ROPE) {
      int f2 = (col & 127) & ~1;           // 2*pair_index within the head
      inv = exp2f((float)f2 * (-kLog2_10000 / 128.0f));
    }
#pragma unroll
    for (int i = 0; i < 8; ++i) {
      const int row = m0 + wave * 16 + hf * 8 + i;
      float val = cv.f[i];
      if (ROPE) {
        float partner = __shfl_xor(val, 1, 32);      // pair element (col ^ 1)
        float pos = (float)(start + (row & (kS - 1)));
        float sn, cs; sincosf(pos * inv, &sn, &cs);
        val = (l15 & 1) ? fmaf(partner, sn, val * cs)    // b' = a*sin + b*cos
                        : fmaf(val, cs, -partner * sn);  // a' = a*cos - b*sin
      }
      C[row * N + col] = val;
    }
  }
}

// ---------------------------------------------------------------------------
// Flash attention. Block = 128 threads = 4 waves; each wave: 16 queries of one
// (b,h). Block cooperatively stages 32-key K/V chunks (bf16) into LDS.
// ---------------------------------------------------------------------------
__global__ __launch_bounds__(128) void attn_kernel(const float* __restrict__ Q,
    const float* __restrict__ Kn, const float* __restrict__ Vn,
    const float* __restrict__ cK, const float* __restrict__ cV,
    float* __restrict__ AO, const int* __restrict__ sp) {
  __shared__ __align__(16) unsigned short sK[32 * 136];    // [key][feat]
  __shared__ __align__(16) unsigned short sV[128 * 40];    // [feat][key]
  __shared__ __align__(16) unsigned short sP[4][16 * 32];  // per-wave [q][key]
  const int t = threadIdx.x;
  const int wave = t >> 5, lane = t & 31;
  const int hf = lane >> 4, l15 = lane & 15;
  const int b = blockIdx.y / kH, h = blockIdx.y % kH, g = h / kNREP;
  const int q0 = blockIdx.x * 64 + wave * 16;
  const int start = sp[0];
  const int T = start + kS;

  // Q fragments straight from global (A-layout, 16x32 per k-step)
  v16bf qf[4];
  {
    const float* qp = Q + ((b * kS + q0 + l15) * kH + h) * kHD;
#pragma unroll
    for (int j = 0; j < 4; ++j) {
      V16 f; int f0 = j * 32 + hf * 8;
#pragma unroll
      for (int e = 0; e < 8; ++e) { f.s[e] = f2bf(qp[f0 + e]); f.s[8 + e] = f2bf(qp[f0 + 16 + e]); }
      qf[j] = f.v;
    }
  }
  v8f o[8] = {};
  float mrow[8], lrow[8];
#pragma unroll
  for (int i = 0; i < 8; ++i) { mrow[i] = -1e30f; lrow[i] = 0.0f; }
  const float scale = 0.088388347648318447f;   // 1/sqrt(128)
  const int nch = (T + 31) >> 5;
  const unsigned short* sK0 = &sK[l15 * 136];
  const unsigned short* sK1 = &sK[(16 + l15) * 136];
  for (int c = 0; c < nch; ++c) {
    const int tb = c * 32;
    __syncthreads();
#pragma unroll
    for (int i = 0; i < 8; ++i) {              // stage 32 keys x 128 feats
      int qq = t + i * 128;
      int key = qq >> 5, fc = (qq & 31) * 4;
      int tt = tb + key;
      float4 kv = make_float4(0.f, 0.f, 0.f, 0.f), vv = kv;
      if (tt < start) {
        int off = ((b * kMAXS + tt) * kKVH + g) * kHD + fc;
        kv = *(const float4*)(cK + off);
        vv = *(const float4*)(cV + off);
      } else if (tt < T) {
        int off = ((b * kS + (tt - start)) * kKVH + g) * kHD + fc;
        kv = *(const float4*)(Kn + off);
        vv = *(const float4*)(Vn + off);
      }
      uint2 u;
      u.x = f2bf2(kv.x, kv.y);
      u.y = f2bf2(kv.z, kv.w);
      *(uint2*)(&sK[key * 136 + fc]) = u;
      sV[(fc + 0) * 40 + key] = f2bf(vv.x);
      sV[(fc + 1) * 40 + key] = f2bf(vv.y);
      sV[(fc + 2) * 40 + key] = f2bf(vv.z);
      sV[(fc + 3) * 40 + key] = f2bf(vv.w);
    }
    __syncthreads();
    // ---- QK^T: preload all 8 K fragments, then 8 back-to-back WMMAs
    V16 kf[8];
#pragma unroll
    for (int j = 0; j < 4; ++j) {
      int ko = j * 32 + hf * 8;
      kf[2 * j + 0].q[0] = *(const uint4*)(sK0 + ko);
      kf[2 * j + 0].q[1] = *(const uint4*)(sK0 + ko + 16);
      kf[2 * j + 1].q[0] = *(const uint4*)(sK1 + ko);
      kf[2 * j + 1].q[1] = *(const uint4*)(sK1 + ko + 16);
    }
    v8f s0v = {}, s1v = {};                    // keys 0..15 / 16..31
#pragma unroll
    for (int j = 0; j < 4; ++j) {
      s0v = wmma_bf16(qf[j], kf[2 * j + 0].v, s0v);
      s1v = wmma_bf16(qf[j], kf[2 * j + 1].v, s1v);
    }
    V8 S0, S1; S0.v = s0v; S1.v = s1v;
    const bool v0 = (tb + l15) < T, v1 = (tb + 16 + l15) < T;
    float cm[8], al[8], rs[8];
#pragma unroll
    for (int i = 0; i < 8; ++i) {
      S0.f[i] = v0 ? S0.f[i] * scale : -1e30f;
      S1.f[i] = v1 ? S1.f[i] * scale : -1e30f;
      cm[i] = fmaxf(S0.f[i], S1.f[i]);
    }
#pragma unroll
    for (int off = 1; off < 16; off <<= 1)
#pragma unroll
      for (int i = 0; i < 8; ++i) cm[i] = fmaxf(cm[i], __shfl_xor(cm[i], off, 32));
#pragma unroll
    for (int i = 0; i < 8; ++i) {
      float mn = fmaxf(mrow[i], cm[i]);
      al[i] = __expf(mrow[i] - mn);
      mrow[i] = mn;
      S0.f[i] = __expf(S0.f[i] - mn);
      S1.f[i] = __expf(S1.f[i] - mn);
      rs[i] = S0.f[i] + S1.f[i];
    }
#pragma unroll
    for (int off = 1; off < 16; off <<= 1)
#pragma unroll
      for (int i = 0; i < 8; ++i) rs[i] += __shfl_xor(rs[i], off, 32);
#pragma unroll
    for (int i = 0; i < 8; ++i) lrow[i] = lrow[i] * al[i] + rs[i];
#pragma unroll
    for (int j = 0; j < 8; ++j)
#pragma unroll
      for (int i = 0; i < 8; ++i) o[j][i] *= al[i];
    // C-layout -> A-layout for P via per-wave LDS tile (same-wave DS ordering)
#pragma unroll
    for (int i = 0; i < 8; ++i) {
      sP[wave][(hf * 8 + i) * 32 + l15]      = f2bf(S0.f[i]);
      sP[wave][(hf * 8 + i) * 32 + 16 + l15] = f2bf(S1.f[i]);
    }
    V16 pf;
    pf.q[0] = *(const uint4*)(&sP[wave][l15 * 32 + hf * 8]);
    pf.q[1] = *(const uint4*)(&sP[wave][l15 * 32 + 16 + hf * 8]);
    // ---- PV: preload all 8 V fragments, then 8 back-to-back WMMAs
    V16 vf[8];
#pragma unroll
    for (int j = 0; j < 8; ++j) {
      vf[j].q[0] = *(const uint4*)(&sV[(j * 16 + l15) * 40 + hf * 8]);
      vf[j].q[1] = *(const uint4*)(&sV[(j * 16 + l15) * 40 + 16 + hf * 8]);
    }
#pragma unroll
    for (int j = 0; j < 8; ++j) o[j] = wmma_bf16(pf.v, vf[j].v, o[j]);
  }
#pragma unroll
  for (int j = 0; j < 8; ++j)
#pragma unroll
    for (int i = 0; i < 8; ++i)
      AO[((b * kS + q0 + hf * 8 + i) * kH + h) * kHD + j * 16 + l15] =
          o[j][i] / lrow[i];
}

// ---------------------------------------------------------------------------
extern "C" void kernel_launch(void* const* d_in, const int* in_sizes, int n_in,
                              void* d_out, int out_size, void* d_ws, size_t ws_size,
                              hipStream_t stream) {
  (void)in_sizes; (void)n_in; (void)out_size; (void)ws_size;
  const float* x  = (const float*)d_in[0];
  const float* wq = (const float*)d_in[1];
  const float* wk = (const float*)d_in[2];
  const float* wv = (const float*)d_in[3];
  const float* wo = (const float*)d_in[4];
  const float* ck = (const float*)d_in[5];
  const float* cv = (const float*)d_in[6];
  const int*   sp = (const int*)d_in[7];
  float* out = (float*)d_out;

  float* Qw = (float*)d_ws;                                   // 64 MB
  float* Kn = Qw + (size_t)kB * kS * kH * kHD;                // 16 MB
  float* Vn = Kn + (size_t)kB * kS * kKVH * kHD;              // 16 MB
  float* AO = Vn + (size_t)kB * kS * kKVH * kHD;              // 64 MB

  const int M = kB * kS;   // 4096
  // Q/K projections with fused RoPE; V plain.
  gemm_xw_bf16<true><<<dim3(kH * kHD / 128, M / 128), 256, 0, stream>>>(
      x, wq, Qw, M, kH * kHD, kD, sp);
  gemm_xw_bf16<true><<<dim3(kKVH * kHD / 128, M / 128), 256, 0, stream>>>(
      x, wk, Kn, M, kKVH * kHD, kD, sp);
  gemm_xw_bf16<false><<<dim3(kKVH * kHD / 128, M / 128), 256, 0, stream>>>(
      x, wv, Vn, M, kKVH * kHD, kD, sp);

  attn_kernel<<<dim3(kS / 64, kB * kH), 128, 0, stream>>>(Qw, Kn, Vn, ck, cv, AO, sp);

  gemm_xw_bf16<false><<<dim3(kD / 128, M / 128), 256, 0, stream>>>(
      AO, wo, out, M, kD, kD, sp);
}
